// TemporalFusionEncoder_1967095021766
// MI455X (gfx1250) — compile-verified, matrix operand order checked by reading, not run on hardware
//
#include <hip/hip_runtime.h>
#include <hip/hip_bf16.h>
#include <math.h>

// ---------------------------------------------------------------------------
// TemporalFusionEncoder for MI455X (gfx1250, wave32, WMMA).
// All heavy GEMMs use v_wmma_f32_16x16x32_bf16 with fp32 accumulation.
// lap (64MB fp32) is converted once to bf16 (32MB) -> L2-resident across the
// 16 timestep reuses (192MB L2). The big Laplacian GEMM stages the shared
// B-operand (catT tiles) into LDS with a double-buffered pipeline so each
// wave issues only its private A loads from global memory.
// ---------------------------------------------------------------------------

typedef __attribute__((ext_vector_type(16))) __bf16 v16bf;
typedef __attribute__((ext_vector_type(8)))  float  v8f;

union FragU { v16bf v; uint4 q[2]; };

// Load a 16x32 bf16 WMMA operand fragment from a row-major (K contiguous)
// matrix. Works for both A and (transposed-stored) B operands.
// Layout per CDNA5 ISA 7.12.2: lane L holds row (L&15); K elements
// base..base+7 (VGPR0..3) and base+16..base+23 (VGPR4..7), base=(L>=16)?8:0.
__device__ __forceinline__ v16bf ld_frag(const __bf16* __restrict__ base,
                                         int ld, int row0, int k0) {
    const int lane = threadIdx.x & 31;
    const __bf16* p = base + (size_t)(row0 + (lane & 15)) * (size_t)ld
                           + (size_t)(k0 + ((lane >> 4) << 3));
    FragU f;
    f.q[0] = *reinterpret_cast<const uint4*>(p);        // K = base..base+7
    f.q[1] = *reinterpret_cast<const uint4*>(p + 16);   // K = base+16..base+23
    return f.v;
}

// Same fragment layout, but from an LDS-resident 16-row x 32-col tile (ld=32).
__device__ __forceinline__ v16bf ld_frag_lds(const __bf16* base, int row0) {
    const int lane = threadIdx.x & 31;
    const __bf16* p = base + (row0 + (lane & 15)) * 32 + ((lane >> 4) << 3);
    FragU f;
    f.q[0] = *reinterpret_cast<const uint4*>(p);
    f.q[1] = *reinterpret_cast<const uint4*>(p + 16);
    return f.v;
}

__device__ __forceinline__ v8f wmma_bf16(v16bf a, v16bf b, v8f c) {
    return __builtin_amdgcn_wmma_f32_16x16x32_bf16(
        /*neg_a=*/false, a, /*neg_b=*/false, b,
        /*c_mod=*/(short)0, c, /*reuse_a=*/false, /*reuse_b=*/false);
}

// ---------------- constants ----------------
#define NN   4096
#define HH   128
#define BB   2
#define TT   16
#define KPAD 160          // per-batch padded feature count (129 -> 160)
#define NCOL 320          // 2 batches * KPAD columns in the big GEMM
#define NH   (NN*HH)      // 524288

// ---------------- prep kernels ----------------
__global__ __launch_bounds__(256) void k_lap_bf16(const float* __restrict__ lap,
                                                  __bf16* __restrict__ lapBF) {
    size_t i = (size_t)blockIdx.x * 256 + threadIdx.x;
    if (i < (size_t)NN * NN) lapBF[i] = (__bf16)lap[i];
}

// WcT[j][f], j<384 -> gc1_w[f][j], j>=384 -> gc2_w[f][j-384]; rows f>=129 = 0
__global__ __launch_bounds__(256) void k_build_wcT(const float* __restrict__ gc1w,
                                                   const float* __restrict__ gc2w,
                                                   __bf16* __restrict__ wcT) {
    int i = blockIdx.x * 256 + threadIdx.x;
    if (i >= 512 * KPAD) return;
    int j = i / KPAD, f = i % KPAD;
    float v = 0.f;
    if (f < 129) v = (j < 384) ? gc1w[f * 384 + j] : gc2w[f * 128 + (j - 384)];
    wcT[i] = (__bf16)v;
}

__global__ __launch_bounds__(256) void k_build_w1T(const float* __restrict__ w1,
                                                   __bf16* __restrict__ w1T) {
    int i = blockIdx.x * 256 + threadIdx.x;
    if (i >= 512 * 384) return;
    int j = i / 384, k = i % 384;
    w1T[i] = (__bf16)w1[k * 512 + j];
}

__global__ __launch_bounds__(256) void k_build_w2T(const float* __restrict__ w2,
                                                   __bf16* __restrict__ w2T) {
    int i = blockIdx.x * 256 + threadIdx.x;
    if (i >= 128 * 512) return;
    int j = i / 512, k = i % 512;
    w2T[i] = (__bf16)w2[k * 128 + j];
}

__global__ __launch_bounds__(256) void k_zero_h(float* __restrict__ h) {
    size_t i = (size_t)blockIdx.x * 256 + threadIdx.x;
    if (i < (size_t)BB * NH) h[i] = 0.f;
}

// ---------------- per-step kernels ----------------
// catT: [NCOL rows][NN k] bf16. Per batch b rows b*KPAD + r:
//  r==0: x[b,t,k,0]; r in 1..128: h[b][k][r-1]; r in 129..131: static (t==0 only); else 0
__global__ __launch_bounds__(256) void k_pack(const float* __restrict__ x,
                                              const float* __restrict__ h,
                                              __bf16* __restrict__ catT, int t) {
    size_t i = (size_t)blockIdx.x * 256 + threadIdx.x;
    if (i >= (size_t)BB * KPAD * NN) return;
    int k  = (int)(i & (NN - 1));
    int rb = (int)(i >> 12);
    int b  = rb / KPAD;
    int r  = rb % KPAD;
    float v = 0.f;
    if (r == 0)                 v = x[(((size_t)b * TT + t) * NN + k) * 4 + 0];
    else if (r <= 128)          v = h[((size_t)b * NN + k) * HH + (r - 1)];
    else if (r <= 131 && t == 0) v = x[(((size_t)b * TT + 0) * NN + k) * 4 + (r - 128)];
    catT[(size_t)rb * NN + k] = (__bf16)v;
}

// ax[4096 x 320] bf16 = lapBF[4096x4096] @ catT^T.
// Block = 8 waves: 128 rows x 64 cols. Shared B tile (64x32 bf16 = 4KB)
// staged into LDS, double-buffered: stage k+1 while computing k.
__global__ __launch_bounds__(256) void k_gemm_lap(const __bf16* __restrict__ lapBF,
                                                  const __bf16* __restrict__ catT,
                                                  __bf16* __restrict__ ax) {
    __shared__ __bf16 btile[2][64 * 32];

    const int wave  = threadIdx.x >> 5;
    const int mBlk  = blockIdx.x / 5;
    const int nG    = blockIdx.x % 5;
    const int mTile = mBlk * 8 + wave;           // 0..255

    // staging role: each thread moves one 16B chunk per stage
    const int tid = threadIdx.x;
    const int sr  = tid >> 2;                    // B row 0..63
    const int sq  = (tid & 3) << 3;              // k offset 0,8,16,24
    const uint4* src0 = reinterpret_cast<const uint4*>(
        catT + (size_t)(nG * 64 + sr) * NN + sq);

    v8f acc0 = {}, acc1 = {}, acc2 = {}, acc3 = {};

    // stage k-tile 0
    *reinterpret_cast<uint4*>(&btile[0][sr * 32 + sq]) = src0[0];
    __syncthreads();

    for (int kt = 0; kt < 128; ++kt) {
        const int cur = kt & 1;
        if (kt + 1 < 128) {
            // stage next tile into the other buffer (write while others read cur)
            *reinterpret_cast<uint4*>(&btile[1 - cur][sr * 32 + sq]) =
                src0[(kt + 1) * 2];              // +16 bf16 elements = 2 uint4
        }
        v16bf a = ld_frag(lapBF, NN, mTile * 16, kt * 32);
        acc0 = wmma_bf16(a, ld_frag_lds(&btile[cur][0],  0), acc0);
        acc1 = wmma_bf16(a, ld_frag_lds(&btile[cur][0], 16), acc1);
        acc2 = wmma_bf16(a, ld_frag_lds(&btile[cur][0], 32), acc2);
        acc3 = wmma_bf16(a, ld_frag_lds(&btile[cur][0], 48), acc3);
        __syncthreads();
    }

    int lane = threadIdx.x & 31, n = lane & 15, mo = (lane >> 4) * 8;
    v8f* accs[4] = {&acc0, &acc1, &acc2, &acc3};
#pragma unroll
    for (int j = 0; j < 4; ++j)
#pragma unroll
        for (int r = 0; r < 8; ++r)
            ax[(size_t)(mTile * 16 + mo + r) * NCOL + (nG * 4 + j) * 16 + n] =
                (__bf16)((*accs[j])[r]);
}

// gates: per batch [4096 x 160] @ WcT^T[160 x 512] -> sigmoid/tanh split
__global__ __launch_bounds__(256) void k_gates(const __bf16* __restrict__ ax,
                                               const __bf16* __restrict__ wcT,
                                               const float* __restrict__ gc1b,
                                               const float* __restrict__ gc2b,
                                               float* __restrict__ conc,
                                               float* __restrict__ cs) {
    int w = blockIdx.x * 8 + (threadIdx.x >> 5);   // 4096 waves
    int b = w >> 11, r = w & 2047;
    int mTile = r >> 3, nG = r & 7;
    v8f acc0 = {}, acc1 = {}, acc2 = {}, acc3 = {};
    for (int kt = 0; kt < 5; ++kt) {
        v16bf a = ld_frag(ax, NCOL, mTile * 16, b * KPAD + kt * 32);
        acc0 = wmma_bf16(a, ld_frag(wcT, KPAD, (nG * 4 + 0) * 16, kt * 32), acc0);
        acc1 = wmma_bf16(a, ld_frag(wcT, KPAD, (nG * 4 + 1) * 16, kt * 32), acc1);
        acc2 = wmma_bf16(a, ld_frag(wcT, KPAD, (nG * 4 + 2) * 16, kt * 32), acc2);
        acc3 = wmma_bf16(a, ld_frag(wcT, KPAD, (nG * 4 + 3) * 16, kt * 32), acc3);
    }
    int lane = threadIdx.x & 31, n = lane & 15, mo = (lane >> 4) * 8;
    v8f* accs[4] = {&acc0, &acc1, &acc2, &acc3};
#pragma unroll
    for (int j = 0; j < 4; ++j) {
        int jj = (nG * 4 + j) * 16 + n;
#pragma unroll
        for (int rr = 0; rr < 8; ++rr) {
            int m = mTile * 16 + mo + rr;
            float v = (*accs[j])[rr];
            if (jj < 384) {
                v = 1.f / (1.f + expf(-(v + gc1b[jj])));
                conc[((size_t)b * NN + m) * 384 + jj] = v;
            } else {
                int j2 = jj - 384;
                v = tanhf(v + gc2b[j2]);
                cs[((size_t)b * NN + m) * HH + j2] = v;
            }
        }
    }
}

// hn = o * tanh(i * cs); i/o gathered from flattened conc (split crosses nodes);
// cell state c is never updated in the reference (stays 0), so f-gate is dead.
__global__ __launch_bounds__(256) void k_newh(const float* __restrict__ conc,
                                              const float* __restrict__ cs,
                                              float* __restrict__ h) {
    size_t i = (size_t)blockIdx.x * 256 + threadIdx.x;
    if (i >= (size_t)BB * NH) return;
    int b = (int)(i >> 19);
    size_t p = i & (NH - 1);
    const float* cb = conc + (size_t)b * NN * 384;
    float ig = cb[(size_t)NH + p];
    float og = cb[(size_t)2 * NH + p];
    float cv = cs[(size_t)b * NH + p];
    h[i] = og * tanhf(ig * cv);
}

// copy static laplacian-mix columns out of ax (computed in step-0 padding cols)
__global__ __launch_bounds__(256) void k_copy_sxa(const __bf16* __restrict__ ax,
                                                  __bf16* __restrict__ sxa) {
    int i = blockIdx.x * 256 + threadIdx.x;
    if (i >= BB * NN * 3) return;
    int b = i / (NN * 3), rem = i % (NN * 3);
    int m = rem / 3, s = rem % 3;
    sxa[i] = ax[(size_t)m * NCOL + b * KPAD + 129 + s];
}

// ---------------- fusion tail ----------------
__global__ __launch_bounds__(256) void k_build_zin(const float* __restrict__ h,
                                                   const __bf16* __restrict__ sxa,
                                                   const float* __restrict__ sw,
                                                   const float* __restrict__ sb,
                                                   const float* __restrict__ se,
                                                   __bf16* __restrict__ zin) {
    size_t i = (size_t)blockIdx.x * 256 + threadIdx.x;
    if (i >= (size_t)BB * NN * 384) return;
    int col = (int)(i % 384);
    size_t row = i / 384;
    int b = (int)(row >> 12), m = (int)(row & (NN - 1));
    float v;
    if (col < 128) {
        v = h[((size_t)b * NN + m) * HH + col];
    } else if (col < 256) {
        int f = col - 128;
        v = sb[f];
        const __bf16* sx = sxa + ((size_t)b * NN + m) * 3;
#pragma unroll
        for (int s = 0; s < 3; ++s) v += (float)sx[s] * sw[s * 128 + f];
    } else {
        v = se[(size_t)m * HH + (col - 256)];
    }
    zin[i] = (__bf16)v;
}

// z1 = elu(zin @ w1 + b1)
__global__ __launch_bounds__(256) void k_gemm1(const __bf16* __restrict__ zin,
                                               const __bf16* __restrict__ w1T,
                                               const float* __restrict__ b1,
                                               float* __restrict__ z1) {
    int w = blockIdx.x * 8 + (threadIdx.x >> 5);   // 4096 waves
    int mTile = w >> 3, nG = w & 7;
    v8f acc0 = {}, acc1 = {}, acc2 = {}, acc3 = {};
    for (int kt = 0; kt < 12; ++kt) {
        v16bf a = ld_frag(zin, 384, mTile * 16, kt * 32);
        acc0 = wmma_bf16(a, ld_frag(w1T, 384, (nG * 4 + 0) * 16, kt * 32), acc0);
        acc1 = wmma_bf16(a, ld_frag(w1T, 384, (nG * 4 + 1) * 16, kt * 32), acc1);
        acc2 = wmma_bf16(a, ld_frag(w1T, 384, (nG * 4 + 2) * 16, kt * 32), acc2);
        acc3 = wmma_bf16(a, ld_frag(w1T, 384, (nG * 4 + 3) * 16, kt * 32), acc3);
    }
    int lane = threadIdx.x & 31, n = lane & 15, mo = (lane >> 4) * 8;
    v8f* accs[4] = {&acc0, &acc1, &acc2, &acc3};
#pragma unroll
    for (int j = 0; j < 4; ++j) {
        int jj = (nG * 4 + j) * 16 + n;
#pragma unroll
        for (int rr = 0; rr < 8; ++rr) {
            int row = mTile * 16 + mo + rr;
            float v = (*accs[j])[rr] + b1[jj];
            v = (v > 0.f) ? v : expm1f(v);
            z1[(size_t)row * 512 + jj] = v;
        }
    }
}

// GroupNorm(8 groups of 64) * gamma + beta -> bf16.  One wave per row;
// each lane owns 16 contiguous values; group = 4 lanes -> shfl_xor reduce.
__global__ __launch_bounds__(256) void k_gnorm(const float* __restrict__ z1,
                                               const float* __restrict__ gamma,
                                               const float* __restrict__ beta,
                                               __bf16* __restrict__ z2) {
    int row = blockIdx.x * 8 + (threadIdx.x >> 5);
    int lane = threadIdx.x & 31;
    const float* p = z1 + (size_t)row * 512 + lane * 16;
    float vals[16], s = 0.f, ss = 0.f;
#pragma unroll
    for (int e = 0; e < 16; ++e) { vals[e] = p[e]; s += vals[e]; ss += vals[e] * vals[e]; }
    s  += __shfl_xor(s, 1, 4);  s  += __shfl_xor(s, 2, 4);
    ss += __shfl_xor(ss, 1, 4); ss += __shfl_xor(ss, 2, 4);
    float mean = s * (1.f / 64.f);
    float var  = ss * (1.f / 64.f) - mean * mean;
    float inv  = rsqrtf(var + 1e-5f);
#pragma unroll
    for (int e = 0; e < 16; ++e) {
        int j = lane * 16 + e;
        z2[(size_t)row * 512 + j] = (__bf16)((vals[e] - mean) * inv * gamma[j] + beta[j]);
    }
}

// out = z2 @ w2 + b2
__global__ __launch_bounds__(256) void k_gemm2(const __bf16* __restrict__ z2,
                                               const __bf16* __restrict__ w2T,
                                               const float* __restrict__ b2,
                                               float* __restrict__ out) {
    int w = blockIdx.x * 8 + (threadIdx.x >> 5);   // 1024 waves
    int mTile = w >> 1, nG = w & 1;
    v8f acc0 = {}, acc1 = {}, acc2 = {}, acc3 = {};
    for (int kt = 0; kt < 16; ++kt) {
        v16bf a = ld_frag(z2, 512, mTile * 16, kt * 32);
        acc0 = wmma_bf16(a, ld_frag(w2T, 512, (nG * 4 + 0) * 16, kt * 32), acc0);
        acc1 = wmma_bf16(a, ld_frag(w2T, 512, (nG * 4 + 1) * 16, kt * 32), acc1);
        acc2 = wmma_bf16(a, ld_frag(w2T, 512, (nG * 4 + 2) * 16, kt * 32), acc2);
        acc3 = wmma_bf16(a, ld_frag(w2T, 512, (nG * 4 + 3) * 16, kt * 32), acc3);
    }
    int lane = threadIdx.x & 31, n = lane & 15, mo = (lane >> 4) * 8;
    v8f* accs[4] = {&acc0, &acc1, &acc2, &acc3};
#pragma unroll
    for (int j = 0; j < 4; ++j) {
        int jj = (nG * 4 + j) * 16 + n;
#pragma unroll
        for (int rr = 0; rr < 8; ++rr) {
            int row = mTile * 16 + mo + rr;
            out[(size_t)row * 128 + jj] = (*accs[j])[rr] + b2[jj];
        }
    }
}

// ---------------------------------------------------------------------------
extern "C" void kernel_launch(void* const* d_in, const int* in_sizes, int n_in,
                              void* d_out, int out_size, void* d_ws, size_t ws_size,
                              hipStream_t stream) {
    (void)in_sizes; (void)n_in; (void)out_size; (void)ws_size;
    const float* x    = (const float*)d_in[0];
    const float* lap  = (const float*)d_in[1];
    const float* se   = (const float*)d_in[2];
    const float* gc1w = (const float*)d_in[3];
    const float* gc1b = (const float*)d_in[4];
    const float* gc2w = (const float*)d_in[5];
    const float* gc2b = (const float*)d_in[6];
    const float* sw   = (const float*)d_in[7];
    const float* sb   = (const float*)d_in[8];
    const float* w1   = (const float*)d_in[9];
    const float* b1   = (const float*)d_in[10];
    const float* gm   = (const float*)d_in[11];
    const float* bt   = (const float*)d_in[12];
    const float* w2   = (const float*)d_in[13];
    const float* b2   = (const float*)d_in[14];
    float* out = (float*)d_out;

    char* ws = (char*)d_ws;
    size_t off = 0;
    auto alloc = [&](size_t bytes) -> char* {
        char* p = ws + off;
        off += (bytes + 255) & ~(size_t)255;
        return p;
    };
    __bf16* lapBF = (__bf16*)alloc((size_t)NN * NN * 2);
    __bf16* catT  = (__bf16*)alloc((size_t)NCOL * NN * 2);
    __bf16* ax    = (__bf16*)alloc((size_t)NN * NCOL * 2);
    float*  conc  = (float*) alloc((size_t)BB * NN * 384 * 4);
    float*  cs    = (float*) alloc((size_t)BB * NN * HH * 4);
    float*  h     = (float*) alloc((size_t)BB * NN * HH * 4);
    __bf16* sxa   = (__bf16*)alloc((size_t)BB * NN * 3 * 2);
    __bf16* wcT   = (__bf16*)alloc((size_t)512 * KPAD * 2);
    __bf16* w1T   = (__bf16*)alloc((size_t)512 * 384 * 2);
    __bf16* w2T   = (__bf16*)alloc((size_t)128 * 512 * 2);
    __bf16* zin   = (__bf16*)alloc((size_t)BB * NN * 384 * 2);
    float*  z1    = (float*) alloc((size_t)BB * NN * 512 * 4);
    __bf16* z2    = (__bf16*)alloc((size_t)BB * NN * 512 * 2);

    // prep
    k_lap_bf16 <<<65536, 256, 0, stream>>>(lap, lapBF);
    k_build_wcT<<<320, 256, 0, stream>>>(gc1w, gc2w, wcT);
    k_build_w1T<<<768, 256, 0, stream>>>(w1, w1T);
    k_build_w2T<<<256, 256, 0, stream>>>(w2, w2T);
    k_zero_h   <<<4096, 256, 0, stream>>>(h);

    // recurrent steps
    for (int t = 0; t < TT; ++t) {
        k_pack    <<<5120, 256, 0, stream>>>(x, h, catT, t);
        k_gemm_lap<<<160, 256, 0, stream>>>(lapBF, catT, ax);
        if (t == 0) k_copy_sxa<<<96, 256, 0, stream>>>(ax, sxa);
        k_gates   <<<512, 256, 0, stream>>>(ax, wcT, gc1b, gc2b, conc, cs);
        k_newh    <<<4096, 256, 0, stream>>>(conc, cs, h);
    }

    // fusion tail
    k_build_zin<<<12288, 256, 0, stream>>>(h, sxa, sw, sb, se, zin);
    k_gemm1    <<<512, 256, 0, stream>>>(zin, w1T, b1, z1);
    k_gnorm    <<<1024, 256, 0, stream>>>(z1, gm, bt, z2);
    k_gemm2    <<<128, 256, 0, stream>>>(z2, w2T, b2, out);
}
